// PointNetSAModuleMSG_12945031430504
// MI455X (gfx1250) — matrix operand order repaired and checked by reading, not compiled, not run on hardware
//
#include <hip/hip_runtime.h>
#include <hip/hip_bf16.h>

typedef __attribute__((ext_vector_type(16))) __bf16 v16bf;
typedef __attribute__((ext_vector_type(8)))  float  v8f;

#define B_    4
#define N_    8192
#define M_    2048
#define C_    64
#define CT_   64
#define CIN1P 96   // 67 channels zero-padded to 3 K-chunks of 32

__device__ __forceinline__ unsigned short f2bf(float f) {
  unsigned int x = __float_as_uint(f);
  unsigned int r = x + 0x7FFFu + ((x >> 16) & 1u);   // round-to-nearest-even
  return (unsigned short)(r >> 16);
}

// ---------------------------------------------------------------------------
// Furthest point sampling: one 1024-thread block per batch; min-dist array in
// LDS (32KB of the 320KB/WGP). Tie-break = lowest index (matches jnp.argmax).
// ---------------------------------------------------------------------------
__global__ __launch_bounds__(1024)
void fps_kernel(const float* __restrict__ coords, int* __restrict__ fpsIdx) {
  __shared__ float mind[N_];
  __shared__ float sval[1024];
  __shared__ int   sidx[1024];
  __shared__ float q[3];
  const int b = blockIdx.x, t = threadIdx.x;
  const float* cb = coords + (size_t)b * N_ * 3;
  if (t == 0) { q[0] = cb[0]; q[1] = cb[1]; q[2] = cb[2]; fpsIdx[b * M_ + 0] = 0; }
  __syncthreads();
  for (int j = t; j < N_; j += 1024) {
    float dx = cb[j * 3 + 0] - q[0], dy = cb[j * 3 + 1] - q[1], dz = cb[j * 3 + 2] - q[2];
    mind[j] = dx * dx + dy * dy + dz * dz;
  }
  __syncthreads();
  for (int i = 1; i < M_; ++i) {
    float bv = -1.0f; int bi = 0x7fffffff;
    for (int j = t; j < N_; j += 1024) { float v = mind[j]; if (v > bv) { bv = v; bi = j; } }
    sval[t] = bv; sidx[t] = bi;
    __syncthreads();
    for (int s = 512; s > 0; s >>= 1) {
      if (t < s) {
        float ov = sval[t + s]; int oi = sidx[t + s];
        if (ov > sval[t] || (ov == sval[t] && oi < sidx[t])) { sval[t] = ov; sidx[t] = oi; }
      }
      __syncthreads();
    }
    if (t == 0) {
      int last = sidx[0];
      fpsIdx[b * M_ + i] = last;
      q[0] = cb[last * 3 + 0]; q[1] = cb[last * 3 + 1]; q[2] = cb[last * 3 + 2];
    }
    __syncthreads();
    if (i + 1 < M_) {
      for (int j = t; j < N_; j += 1024) {
        float dx = cb[j * 3 + 0] - q[0], dy = cb[j * 3 + 1] - q[1], dz = cb[j * 3 + 2] - q[2];
        float d = dx * dx + dy * dy + dz * dz;
        if (d < mind[j]) mind[j] = d;
      }
      __syncthreads();
    }
  }
}

// ---------------------------------------------------------------------------
// Gather query coordinates (first output block).
// ---------------------------------------------------------------------------
__global__ void queries_kernel(const float* __restrict__ coords,
                               const int* __restrict__ fpsIdx,
                               float* __restrict__ qOut) {
  int g = blockIdx.x * blockDim.x + threadIdx.x;     // over B*M
  if (g >= B_ * M_) return;
  int b = g / M_;
  int j = fpsIdx[g];
  const float* cb = coords + (size_t)b * N_ * 3 + (size_t)j * 3;
  qOut[g * 3 + 0] = cb[0]; qOut[g * 3 + 1] = cb[1]; qOut[g * 3 + 2] = cb[2];
}

// ---------------------------------------------------------------------------
// Ball query: one wave32 per query. Ballot/popc compaction picks the first K
// indices in ascending order (== reference top-k of negated indices), pads
// the remainder with the first hit.
// ---------------------------------------------------------------------------
template <int K>
__global__ __launch_bounds__(256)
void ballquery_kernel(const float* __restrict__ coords, const float* __restrict__ qOut,
                      int* __restrict__ idxOut, float r2) {
  int wv   = (blockIdx.x * blockDim.x + threadIdx.x) >> 5;
  int lane = threadIdx.x & 31;
  if (wv >= B_ * M_) return;
  int b = wv / M_;
  float qx = qOut[wv * 3 + 0], qy = qOut[wv * 3 + 1], qz = qOut[wv * 3 + 2];
  const float* cb = coords + (size_t)b * N_ * 3;
  int* out = idxOut + (size_t)wv * K;
  int cnt = 0, first = 0;
  for (int base = 0; base < N_; base += 32) {
    int j = base + lane;
    float dx = cb[j * 3 + 0] - qx, dy = cb[j * 3 + 1] - qy, dz = cb[j * 3 + 2] - qz;
    bool pred = (dx * dx + dy * dy + dz * dz) <= r2;
    unsigned int m32 = (unsigned int)__ballot(pred);
    if (m32) {
      if (cnt == 0) first = base + (__ffs(m32) - 1);
      int pos = __popc(m32 & ((1u << lane) - 1u));
      if (pred && (cnt + pos) < K) out[cnt + pos] = j;
      cnt += __popc(m32);
      if (cnt >= K) break;
    }
  }
  for (int k = cnt + lane; k < K; k += 32) out[k] = first;
}

// ---------------------------------------------------------------------------
// fp32 -> bf16 weight conversion with zero padding of cin.
// ---------------------------------------------------------------------------
__global__ void convw_kernel(const float* __restrict__ src, unsigned short* __restrict__ dst,
                             int cout, int cin, int cinp) {
  int g = blockIdx.x * blockDim.x + threadIdx.x;
  if (g >= cout * cinp) return;
  int m = g / cinp, k = g % cinp;
  dst[g] = (k < cin) ? f2bf(src[m * cin + k]) : (unsigned short)0;
}

// ---------------------------------------------------------------------------
// Grouped 2-layer MLP + max-over-K, WMMA bf16 (f32 accumulate).
// Block = 128 threads (4 waves), processes 64 columns = 64/K queries.
// LDS: X (64 x 96 bf16, col-major) + H1 (64 x C1 bf16, col-major).
// ---------------------------------------------------------------------------
template <int K, int C1, int C2>
__global__ __launch_bounds__(128)
void mlp_kernel(const float* __restrict__ coords,
                const float* __restrict__ features,
                const float* __restrict__ qOut,
                const int*   __restrict__ idxBuf,
                const unsigned int* __restrict__ W1,   // bf16 [C1][96] as uints
                const float* __restrict__ b1,
                const unsigned int* __restrict__ W2,   // bf16 [C2][C1] as uints
                const float* __restrict__ b2,
                float* __restrict__ fOut, int chOff) {
  constexpr int TQ  = 64 / K;   // queries per block
  constexpr int TPQ = K / 16;   // 16-column tiles per query
  __shared__ unsigned int sX[64 * CIN1P / 2];
  __shared__ unsigned int sH1[64 * C1 / 2];

  const int tid   = threadIdx.x;
  const int lane  = tid & 31;
  const int wave  = tid >> 5;
  const int b     = blockIdx.x / (M_ / TQ);
  const int qbase = (blockIdx.x % (M_ / TQ)) * TQ;

  // ---- gather: X[col][0..2]=rel coords, [3..66]=features, [67..95]=0 (bf16)
  {
    unsigned short* sXs = (unsigned short*)sX;
    const float* fb = features + (size_t)b * C_ * N_;
    for (int e = tid; e < 64 * CIN1P; e += 128) {
      int col = e / CIN1P, r = e % CIN1P;
      int q = qbase + col / K;
      int k = col % K;
      int idx = idxBuf[((size_t)(b * M_ + q)) * K + k];
      float v;
      if (r < 3)       v = coords[((size_t)b * N_ + idx) * 3 + r] - qOut[((size_t)(b * M_ + q)) * 3 + r];
      else if (r < 67) v = fb[(size_t)(r - 3) * N_ + idx];
      else             v = 0.f;
      sXs[col * CIN1P + r] = f2bf(v);
    }
  }
  __syncthreads();

  // ---- layer 1: H1 = relu(W1 * X + b1)
  {
    unsigned short* sH1s = (unsigned short*)sH1;
    const int nT = (C1 / 16) * 4;
    for (int task = wave; task < nT; task += 4) {
      int ct = task >> 2, colt = task & 3;
      int mrow  = ct * 16 + (lane & 15);
      int ncol  = colt * 16 + (lane & 15);
      int koffA = (lane < 16) ? 0 : 8;
      int koffB = (lane < 16) ? 0 : 8;
      v8f acc = {};
      #pragma unroll
      for (int kc = 0; kc < CIN1P / 32; ++kc) {
        union { v16bf v; unsigned int u[8]; } A, Bf;
        const unsigned int* pa0 = W1 + (mrow * CIN1P + kc * 32 + koffA) / 2;
        const unsigned int* pa1 = W1 + (mrow * CIN1P + kc * 32 + 16 + koffA) / 2;
        #pragma unroll
        for (int i = 0; i < 4; ++i) { A.u[i] = pa0[i]; A.u[4 + i] = pa1[i]; }
        const unsigned int* pb = sX + (ncol * CIN1P + kc * 32 + koffB) / 2;
        #pragma unroll
        for (int i = 0; i < 8; ++i) Bf.u[i] = pb[(i & 3) + ((i >> 2) << 3)];
        acc = __builtin_amdgcn_wmma_f32_16x16x32_bf16(false, A.v, false, Bf.v,
                                                      (short)0, acc, false, false);
      }
      int mbase = ct * 16 + ((lane >= 16) ? 8 : 0);
      int col   = colt * 16 + (lane & 15);
      #pragma unroll
      for (int v = 0; v < 8; ++v) {
        float x = fmaxf(acc[v] + b1[mbase + v], 0.f);
        sH1s[col * C1 + mbase + v] = f2bf(x);
      }
    }
  }
  __syncthreads();

  // ---- layer 2 + max over K
  for (int ct = wave; ct < C2 / 16; ct += 4) {
    int mrow  = ct * 16 + (lane & 15);
    int koffA = (lane < 16) ? 0 : 8;
    int koffB = (lane < 16) ? 0 : 8;
    int mbase = ct * 16 + ((lane >= 16) ? 8 : 0);
    for (int ql = 0; ql < TQ; ++ql) {
      v8f vmax = {};                       // relu outputs are >= 0
      #pragma unroll
      for (int t2 = 0; t2 < TPQ; ++t2) {
        int colt = ql * TPQ + t2;
        int ncol = colt * 16 + (lane & 15);
        v8f acc = {};
        #pragma unroll
        for (int kc = 0; kc < C1 / 32; ++kc) {
          union { v16bf v; unsigned int u[8]; } A, Bf;
          const unsigned int* pa0 = W2 + (mrow * C1 + kc * 32 + koffA) / 2;
          const unsigned int* pa1 = W2 + (mrow * C1 + kc * 32 + 16 + koffA) / 2;
          #pragma unroll
          for (int i = 0; i < 4; ++i) { A.u[i] = pa0[i]; A.u[4 + i] = pa1[i]; }
          const unsigned int* pb = sH1 + (ncol * C1 + kc * 32 + koffB) / 2;
          #pragma unroll
          for (int i = 0; i < 8; ++i) Bf.u[i] = pb[(i & 3) + ((i >> 2) << 3)];
          acc = __builtin_amdgcn_wmma_f32_16x16x32_bf16(false, A.v, false, Bf.v,
                                                        (short)0, acc, false, false);
        }
        #pragma unroll
        for (int v = 0; v < 8; ++v) {
          float x = fmaxf(acc[v] + b2[mbase + v], 0.f);
          vmax[v] = fmaxf(vmax[v], x);
        }
      }
      // max across the 16 columns of the tile (stays within 16-lane halves)
      #pragma unroll
      for (int d = 1; d < 16; d <<= 1) {
        #pragma unroll
        for (int v = 0; v < 8; ++v) vmax[v] = fmaxf(vmax[v], __shfl_xor(vmax[v], d, 32));
      }
      if ((lane & 15) == 0) {
        int q = qbase + ql;
        float* o = fOut + (size_t)b * 512 * M_ + (size_t)chOff * M_ + q;
        #pragma unroll
        for (int v = 0; v < 8; ++v) o[(size_t)(mbase + v) * M_] = vmax[v];
      }
    }
  }
}

// ---------------------------------------------------------------------------
// t_embed gather + max over K (third output block). L2-resident gathers.
// ---------------------------------------------------------------------------
template <int K>
__global__ void tmax_kernel(const float* __restrict__ t_embed, const int* __restrict__ idxBuf,
                            float* __restrict__ tOut, int scaleOff) {
  int g = blockIdx.x * blockDim.x + threadIdx.x;     // over B*M*CT
  if (g >= B_ * M_ * CT_) return;
  int c = g % CT_;
  int bm = g / CT_;
  int b = bm / M_, m = bm % M_;
  const float* tb = t_embed + (size_t)b * CT_ * N_ + (size_t)c * N_;
  const int* idx = idxBuf + (size_t)bm * K;
  float v = -INFINITY;
  for (int k = 0; k < K; ++k) v = fmaxf(v, tb[idx[k]]);
  tOut[(size_t)b * 192 * M_ + (size_t)(scaleOff + c) * M_ + m] = v;
}

// ---------------------------------------------------------------------------
extern "C" void kernel_launch(void* const* d_in, const int* in_sizes, int n_in,
                              void* d_out, int out_size, void* d_ws, size_t ws_size,
                              hipStream_t stream) {
  (void)in_sizes; (void)n_in; (void)out_size; (void)ws_size;
  const float* coords   = (const float*)d_in[0];
  const float* features = (const float*)d_in[1];
  const float* t_embed  = (const float*)d_in[2];
  // params flat order: s0: W1[3] b1[4] W2[5] b2[6]; s1: 7..10; s2: 11..14

  char* ws = (char*)d_ws;
  size_t off = 0;
  auto alloc = [&](size_t bytes) -> char* {
    char* p = ws + off; off += (bytes + 255) & ~(size_t)255; return p;
  };
  int* fpsIdx = (int*)alloc((size_t)B_ * M_ * 4);
  int* idx0   = (int*)alloc((size_t)B_ * M_ * 16 * 4);
  int* idx1   = (int*)alloc((size_t)B_ * M_ * 32 * 4);
  int* idx2   = (int*)alloc((size_t)B_ * M_ * 64 * 4);
  unsigned short* W1b0 = (unsigned short*)alloc(64 * 96 * 2);
  unsigned short* W2b0 = (unsigned short*)alloc(128 * 64 * 2);
  unsigned short* W1b1 = (unsigned short*)alloc(128 * 96 * 2);
  unsigned short* W2b1 = (unsigned short*)alloc(128 * 128 * 2);
  unsigned short* W1b2 = (unsigned short*)alloc(128 * 96 * 2);
  unsigned short* W2b2 = (unsigned short*)alloc(256 * 128 * 2);

  float* qOut = (float*)d_out;                       // (B, M, 3)
  float* fOut = qOut + (size_t)B_ * M_ * 3;          // (B, 512, M)
  float* tOut = fOut + (size_t)B_ * 512 * M_;        // (B, 192, M)

  fps_kernel<<<B_, 1024, 0, stream>>>(coords, fpsIdx);
  queries_kernel<<<(B_ * M_ + 255) / 256, 256, 0, stream>>>(coords, fpsIdx, qOut);

  ballquery_kernel<16><<<B_ * M_ / 8, 256, 0, stream>>>(coords, qOut, idx0, 0.1f * 0.1f);
  ballquery_kernel<32><<<B_ * M_ / 8, 256, 0, stream>>>(coords, qOut, idx1, 0.2f * 0.2f);
  ballquery_kernel<64><<<B_ * M_ / 8, 256, 0, stream>>>(coords, qOut, idx2, 0.4f * 0.4f);

  convw_kernel<<<(64 * 96 + 255) / 256, 256, 0, stream>>>((const float*)d_in[3],  W1b0,  64,  67, 96);
  convw_kernel<<<(128 * 64 + 255) / 256, 256, 0, stream>>>((const float*)d_in[5],  W2b0, 128,  64, 64);
  convw_kernel<<<(128 * 96 + 255) / 256, 256, 0, stream>>>((const float*)d_in[7],  W1b1, 128,  67, 96);
  convw_kernel<<<(128 * 128 + 255) / 256, 256, 0, stream>>>((const float*)d_in[9], W2b1, 128, 128, 128);
  convw_kernel<<<(128 * 96 + 255) / 256, 256, 0, stream>>>((const float*)d_in[11], W1b2, 128,  67, 96);
  convw_kernel<<<(256 * 128 + 255) / 256, 256, 0, stream>>>((const float*)d_in[13], W2b2, 256, 128, 128);

  mlp_kernel<16, 64, 128><<<B_ * M_ / 4, 128, 0, stream>>>(
      coords, features, qOut, idx0,
      (const unsigned int*)W1b0, (const float*)d_in[4],
      (const unsigned int*)W2b0, (const float*)d_in[6], fOut, 0);
  mlp_kernel<32, 128, 128><<<B_ * M_ / 2, 128, 0, stream>>>(
      coords, features, qOut, idx1,
      (const unsigned int*)W1b1, (const float*)d_in[8],
      (const unsigned int*)W2b1, (const float*)d_in[10], fOut, 128);
  mlp_kernel<64, 128, 256><<<B_ * M_, 128, 0, stream>>>(
      coords, features, qOut, idx2,
      (const unsigned int*)W1b2, (const float*)d_in[12],
      (const unsigned int*)W2b2, (const float*)d_in[14], fOut, 256);

  tmax_kernel<16><<<(B_ * M_ * CT_ + 255) / 256, 256, 0, stream>>>(t_embed, idx0, tOut, 0);
  tmax_kernel<32><<<(B_ * M_ * CT_ + 255) / 256, 256, 0, stream>>>(t_embed, idx1, tOut, 64);
  tmax_kernel<64><<<(B_ * M_ * CT_ + 255) / 256, 256, 0, stream>>>(t_embed, idx2, tOut, 128);
}
